// TransformerAutoRegressiveDecoder_68487548502339
// MI455X (gfx1250) — compile-verified, matrix operand order checked by reading, not run on hardware
//
#include <hip/hip_runtime.h>
#include <hip/hip_bf16.h>
#include <math.h>

#define B_   16
#define T_   48
#define D_   256
#define H_   8
#define HD_  32
#define L_   3
#define OUT_ 6

typedef __attribute__((ext_vector_type(16))) _Float16 v16h;
typedef __attribute__((ext_vector_type(8)))  _Float16 h8;
typedef __attribute__((ext_vector_type(8)))  float    v8f;

__device__ __forceinline__ float gelu_exact(float x) {
    return 0.5f * x * (1.0f + erff(x * 0.70710678118654752440f));
}

// ---------------------------------------------------------------------------
// WMMA fragment helpers (CDNA5 v_wmma_f32_16x16x32_f16, wave32)
// A 16x32 f16: lane m=lane&15, half=lane>>4; elems e<8 -> K=half*8+e,
//              e>=8 -> K=16+half*8+(e-8)  (CDNA5 ISA 7.12.2)
// B 32x16 f16: mirrored with lane -> N. For y = x @ W^T with W row-major
// [N,K], B(k,n)=W[n,k] so the B load pattern equals the A load pattern.
// C/D 16x16 f32: VGPR j -> M=j (lanes 0-15) / M=j+8 (lanes 16-31), lane->N.
// ---------------------------------------------------------------------------
__device__ __forceinline__ v16h load_a_frag_f32(const float* X, int ld, int kbase, int lane) {
    int m = lane & 15, half = lane >> 4;
    const float* p = X + m * ld + kbase + half * 8;
    v16h r;
#pragma unroll
    for (int j = 0; j < 8; ++j) {
        r[j]     = (_Float16)p[j];
        r[8 + j] = (_Float16)p[16 + j];
    }
    return r;
}

__device__ __forceinline__ v16h load_b_frag_f16(const _Float16* W, int ldk, int nbase, int kbase, int lane) {
    int n = lane & 15, half = lane >> 4;
    const _Float16* p = W + (size_t)(nbase + n) * ldk + kbase + half * 8;
    h8 lo = *reinterpret_cast<const h8*>(p);
    h8 hi = *reinterpret_cast<const h8*>(p + 16);
    v16h r;
#pragma unroll
    for (int j = 0; j < 8; ++j) { r[j] = lo[j]; r[8 + j] = hi[j]; }
    return r;
}

// Team matmul: D[16 x 8*NTPW*16] = A[16 x KT*32] @ W^T (+bias, opt gelu)
// A in LDS fp32 (row stride lda), W f16 row-major [N, ldk], D to LDS fp32.
// 8 waves; wave w owns n-tiles {w, w+8, ..., w+8*(NTPW-1)}. Accumulators
// stay in VGPRs across the whole K loop; B fragments are preloaded in
// chunks of 4 tiles so 8 global_load_b128 are in flight per WMMA burst
// (software pipelining -- avoids a full s_wait_loadcnt 0 per WMMA).
template <int NTPW, int KT, bool GELU>
__device__ __forceinline__ void mm_wmma(const float* Alds, int lda,
                                        const _Float16* W, int ldk,
                                        const float* bias,
                                        float* Dlds, int ldd, int tid) {
    int wave = tid >> 5, lane = tid & 31;
    int n = lane & 15, half = lane >> 4;
    v8f acc[NTPW];
#pragma unroll
    for (int i = 0; i < NTPW; ++i) acc[i] = {};

    constexpr int KC = (KT < 4) ? KT : 4;  // k-tile chunk
#pragma unroll
    for (int kc = 0; kc < KT; kc += KC) {
        v16h a[KC];
#pragma unroll
        for (int k = 0; k < KC; ++k)
            a[k] = load_a_frag_f32(Alds, lda, (kc + k) * 32, lane);
#pragma unroll
        for (int i = 0; i < NTPW; ++i) {
            int nt = wave + i * 8;
            v16h b[KC];
#pragma unroll
            for (int k = 0; k < KC; ++k)
                b[k] = load_b_frag_f16(W, ldk, nt * 16, (kc + k) * 32, lane);
#pragma unroll
            for (int k = 0; k < KC; ++k)
                acc[i] = __builtin_amdgcn_wmma_f32_16x16x32_f16(
                    false, a[k], false, b[k], (short)0, acc[i], false, false);
        }
    }
#pragma unroll
    for (int i = 0; i < NTPW; ++i) {
        int nt = wave + i * 8;
        int nc = nt * 16 + n;
        float bb = bias[nc];
#pragma unroll
        for (int j = 0; j < 8; ++j) {
            int m = j + 8 * half;
            float v = acc[i][j] + bb;
            if (GELU) v = gelu_exact(v);
            Dlds[m * ldd + nc] = v;
        }
    }
}

// out = LayerNorm(A + Badd) rowwise over D_=256, eps 1e-5.
// 16 lanes per row (tid>>4 = row, each lane owns 16 contiguous cols);
// reduction via wave32 shfl_xor within the aligned 16-lane group.
__device__ __forceinline__ void ln_residual(const float* A, const float* Badd,
                                            const float* g, const float* bet,
                                            float* outp, int tid) {
    int row = tid >> 4, c0 = (tid & 15) << 4;
    float vals[16];
    float s = 0.f, s2 = 0.f;
#pragma unroll
    for (int i = 0; i < 16; ++i) {
        float v = A[row * D_ + c0 + i] + Badd[row * D_ + c0 + i];
        vals[i] = v; s += v; s2 += v * v;
    }
#pragma unroll
    for (int off = 1; off < 16; off <<= 1) {
        s  += __shfl_xor(s,  off, 32);
        s2 += __shfl_xor(s2, off, 32);
    }
    float mean = s * (1.0f / 256.0f);
    float var  = s2 * (1.0f / 256.0f) - mean * mean;
    float inv  = rsqrtf(var + 1e-5f);
#pragma unroll
    for (int i = 0; i < 16; ++i) {
        int d = c0 + i;
        outp[row * D_ + d] = (vals[i] - mean) * inv * g[d] + bet[d];
    }
}

// ---------------------------------------------------------------------------
// Kernel 1: fp32 -> f16 weight conversion (weights then stream from L2)
// ---------------------------------------------------------------------------
__global__ void cvt_f32_to_f16(const float* __restrict__ src,
                               _Float16* __restrict__ dst, int n) {
    int i = blockIdx.x * blockDim.x + threadIdx.x;
    if (i < n) dst[i] = (_Float16)src[i];
}

// ---------------------------------------------------------------------------
// Kernel 2: memory MLP + cross-attention constants.
// All cross-attn keys are identical (memory broadcast) -> softmax uniform ->
// cross-attn output == o_w @ (W_v @ mem + b_v) + o_b, independent of Q.
// ---------------------------------------------------------------------------
__global__ void __launch_bounds__(256) setup_memory_kernel(
    const float* z_style, const float* z_skill,
    const float* mem_w1, const float* mem_b1,
    const float* mem_ln_g, const float* mem_ln_b,
    const float* mem_w2, const float* mem_b2,
    const float* ca_qkv_w, const float* ca_qkv_b,
    const float* ca_o_w, const float* ca_o_b,
    float* ca_const /* [L][B][D] */) {
    __shared__ float sA[B_][D_];
    __shared__ float sB[B_][D_];
    int tid = threadIdx.x;
    int d = tid;
    // t1 = cond @ mem_w1.T + mem_b1   (cond = [z_style | z_skill], 128)
    for (int b = 0; b < B_; ++b) {
        float acc = mem_b1[d];
        const float* wr = mem_w1 + d * 128;
        for (int k = 0; k < 64; ++k) acc += wr[k] * z_style[b * 64 + k];
        for (int k = 0; k < 64; ++k) acc += wr[64 + k] * z_skill[b * 64 + k];
        sA[b][d] = acc;
    }
    __syncthreads();
    // h = gelu(LN(t1))
    {
        int row = tid >> 4, c0 = (tid & 15) << 4;
        float vals[16], s = 0.f, s2 = 0.f;
#pragma unroll
        for (int i = 0; i < 16; ++i) {
            float v = sA[row][c0 + i];
            vals[i] = v; s += v; s2 += v * v;
        }
#pragma unroll
        for (int off = 1; off < 16; off <<= 1) {
            s += __shfl_xor(s, off, 32); s2 += __shfl_xor(s2, off, 32);
        }
        float mean = s * (1.0f / 256.0f);
        float var  = s2 * (1.0f / 256.0f) - mean * mean;
        float inv  = rsqrtf(var + 1e-5f);
#pragma unroll
        for (int i = 0; i < 16; ++i) {
            int dd = c0 + i;
            sB[row][dd] = gelu_exact((vals[i] - mean) * inv * mem_ln_g[dd] + mem_ln_b[dd]);
        }
    }
    __syncthreads();
    // mem_vec = h @ mem_w2.T + mem_b2
    for (int b = 0; b < B_; ++b) {
        float acc = mem_b2[d];
        const float* wr = mem_w2 + d * 256;
        for (int k = 0; k < 256; ++k) acc += wr[k] * sB[b][k];
        sA[b][d] = acc;
    }
    __syncthreads();
    for (int l = 0; l < L_; ++l) {
        // v = mem_vec @ Wv.T + bv   (rows 512..767 of ca_qkv_w[l])
        for (int b = 0; b < B_; ++b) {
            float acc = ca_qkv_b[l * 768 + 512 + d];
            const float* wr = ca_qkv_w + ((size_t)l * 768 + 512 + d) * 256;
            for (int k = 0; k < 256; ++k) acc += wr[k] * sA[b][k];
            sB[b][d] = acc;
        }
        __syncthreads();
        // ca_const = v @ o_w.T + o_b
        for (int b = 0; b < B_; ++b) {
            float acc = ca_o_b[l * 256 + d];
            const float* wr = ca_o_w + ((size_t)l * 256 + d) * 256;
            for (int k = 0; k < 256; ++k) acc += wr[k] * sB[b][k];
            ca_const[((size_t)l * B_ + b) * D_ + d] = acc;
        }
        __syncthreads();
    }
}

// ---------------------------------------------------------------------------
// Kernel 3: persistent single-workgroup incremental decoder (48 steps).
// 256 threads = 8 wave32s on one WGP; activations in LDS; KV cache in ws.
// ---------------------------------------------------------------------------
__global__ void __launch_bounds__(256) decoder_kernel(
    const float* start_token, const float* emb_w, const float* emb_b,
    const _Float16* w_qkv16, const float* sa_qkv_b,
    const _Float16* w_o16,   const float* sa_o_b,
    const _Float16* w_ff1,   const float* ff1_b,
    const _Float16* w_ff2,   const float* ff2_b,
    const float* ln1_g, const float* ln1_b,
    const float* ln2_g, const float* ln2_b,
    const float* ln3_g, const float* ln3_b,
    const float* ca_const,
    const float* op_w1, const float* op_b1,
    const float* op_w2, const float* op_b2,
    float* kcache, float* vcache, float* out) {
    __shared__ float s_x[B_][D_];        // activations
    __shared__ float s_big[B_][1024];    // qkv (768 used) / ffn hidden / head
    __shared__ float s_attn[B_][D_];     // attention output
    __shared__ float s_tmp[B_][D_];      // residual branch
    __shared__ float s_sc[128][T_ + 1];  // attention scores per (b,h)
    __shared__ float s_in[B_][8];        // current input token (OUT_=6 used)

    int tid = threadIdx.x;

    if (tid < B_ * OUT_) {
        int b = tid / OUT_, o = tid % OUT_;
        s_in[b][o] = start_token[o];
    }
    __syncthreads();

    for (int t = 0; t < T_; ++t) {
        // ---- embedding + sinusoidal PE (column d = tid) ----
        {
            int d = tid;
            int i2 = d >> 1;
            float ang = (float)t * expf(-((float)(2 * i2)) * (9.210340371976184f / 256.0f));
            float pe = (d & 1) ? cosf(ang) : sinf(ang);
            float eb = emb_b[d] + pe;
            for (int b = 0; b < B_; ++b) {
                float acc = eb;
#pragma unroll
                for (int o = 0; o < OUT_; ++o) acc += s_in[b][o] * emb_w[d * OUT_ + o];
                s_x[b][d] = acc;
            }
        }
        __syncthreads();

        for (int l = 0; l < L_; ++l) {
            // ---- self-attn QKV: [16,768] = x @ Wqkv^T + b ----
            mm_wmma<6, 8, false>(&s_x[0][0], D_, w_qkv16 + (size_t)l * 768 * 256, 256,
                                 sa_qkv_b + l * 768, &s_big[0][0], 1024, tid);
            __syncthreads();
            // append K,V at position t
            for (int i = 0; i < 16; ++i) {
                int e = tid + (i << 8);
                int b = e >> 8, d = e & 255;
                size_t idx = (((size_t)l * B_ + b) * T_ + t) * D_ + d;
                kcache[idx] = s_big[b][256 + d];
                vcache[idx] = s_big[b][512 + d];
            }
            __threadfence_block();
            __syncthreads();
            // ---- attention over cached keys 0..t (thread = one (b,h)) ----
            if (tid < 128) {
                int b = tid >> 3, h = tid & 7;
                const float4* q4 = reinterpret_cast<const float4*>(&s_big[b][h * HD_]);
                float4 qv[8];
#pragma unroll
                for (int c = 0; c < 8; ++c) qv[c] = q4[c];
                const float scale = 0.17677669529663688f;  // 1/sqrt(32)
                size_t base = ((size_t)l * B_ + b) * T_ * D_ + h * HD_;
                for (int j = 0; j <= t; ++j) {
                    const float4* kp = reinterpret_cast<const float4*>(kcache + base + (size_t)j * D_);
                    float dacc = 0.f;
#pragma unroll
                    for (int c = 0; c < 8; ++c) {
                        float4 kk = kp[c];
                        dacc += qv[c].x * kk.x + qv[c].y * kk.y + qv[c].z * kk.z + qv[c].w * kk.w;
                    }
                    s_sc[tid][j] = dacc * scale;
                }
                float mx = -1e30f;
                for (int j = 0; j <= t; ++j) mx = fmaxf(mx, s_sc[tid][j]);
                float4 ov[8];
#pragma unroll
                for (int c = 0; c < 8; ++c) ov[c] = make_float4(0.f, 0.f, 0.f, 0.f);
                float ssum = 0.f;
                for (int j = 0; j <= t; ++j) {
                    float p = __expf(s_sc[tid][j] - mx);
                    ssum += p;
                    const float4* vp = reinterpret_cast<const float4*>(vcache + base + (size_t)j * D_);
#pragma unroll
                    for (int c = 0; c < 8; ++c) {
                        float4 vv = vp[c];
                        ov[c].x += p * vv.x; ov[c].y += p * vv.y;
                        ov[c].z += p * vv.z; ov[c].w += p * vv.w;
                    }
                }
                float rs = 1.0f / ssum;
                float4* oo = reinterpret_cast<float4*>(&s_attn[b][h * HD_]);
#pragma unroll
                for (int c = 0; c < 8; ++c)
                    oo[c] = make_float4(ov[c].x * rs, ov[c].y * rs, ov[c].z * rs, ov[c].w * rs);
            }
            __syncthreads();
            // ---- out projection + residual + LN1 ----
            mm_wmma<2, 8, false>(&s_attn[0][0], D_, w_o16 + (size_t)l * 256 * 256, 256,
                                 sa_o_b + l * 256, &s_tmp[0][0], D_, tid);
            __syncthreads();
            ln_residual(&s_x[0][0], &s_tmp[0][0], ln1_g + l * 256, ln1_b + l * 256, &s_x[0][0], tid);
            __syncthreads();
            // ---- cross-attn constant + LN2 ----
            for (int i = 0; i < 16; ++i) {
                int e = tid + (i << 8);
                int b = e >> 8, d = e & 255;
                s_tmp[b][d] = ca_const[((size_t)l * B_ + b) * D_ + d];
            }
            __syncthreads();
            ln_residual(&s_x[0][0], &s_tmp[0][0], ln2_g + l * 256, ln2_b + l * 256, &s_x[0][0], tid);
            __syncthreads();
            // ---- FFN: gelu(x@W1^T+b1)@W2^T+b2, residual + LN3 ----
            mm_wmma<8, 8, true>(&s_x[0][0], D_, w_ff1 + (size_t)l * 1024 * 256, 256,
                                ff1_b + l * 1024, &s_big[0][0], 1024, tid);
            __syncthreads();
            mm_wmma<2, 32, false>(&s_big[0][0], 1024, w_ff2 + (size_t)l * 256 * 1024, 1024,
                                  ff2_b + l * 256, &s_tmp[0][0], D_, tid);
            __syncthreads();
            ln_residual(&s_x[0][0], &s_tmp[0][0], ln3_g + l * 256, ln3_b + l * 256, &s_x[0][0], tid);
            __syncthreads();
        }

        // ---- output head: out_t = gelu(x@op_w1^T+b1)@op_w2^T+b2 ----
        for (int i = 0; i < 8; ++i) {
            int e = tid + (i << 8);  // 0..2047
            int b = e >> 7, j = e & 127;
            float acc = op_b1[j];
            const float* wr = op_w1 + j * 256;
            for (int k = 0; k < 256; ++k) acc += wr[k] * s_x[b][k];
            s_big[b][j] = gelu_exact(acc);
        }
        __syncthreads();
        if (tid < B_ * OUT_) {
            int b = tid / OUT_, o = tid % OUT_;
            float acc = op_b2[o];
            const float* wr = op_w2 + o * 128;
            for (int k = 0; k < 128; ++k) acc += wr[k] * s_big[b][k];
            out[((size_t)b * T_ + t) * OUT_ + o] = acc;
            s_in[b][o] = acc;  // token for step t+1
        }
        __syncthreads();
    }
}

// ---------------------------------------------------------------------------
extern "C" void kernel_launch(void* const* d_in, const int* in_sizes, int n_in,
                              void* d_out, int out_size, void* d_ws, size_t ws_size,
                              hipStream_t stream) {
    (void)in_sizes; (void)n_in; (void)out_size; (void)ws_size;
    const float* z_style  = (const float*)d_in[0];
    const float* z_skill  = (const float*)d_in[1];
    const float* start_tk = (const float*)d_in[2];
    const float* mem_w1   = (const float*)d_in[3];
    const float* mem_b1   = (const float*)d_in[4];
    const float* mem_ln_g = (const float*)d_in[5];
    const float* mem_ln_b = (const float*)d_in[6];
    const float* mem_w2   = (const float*)d_in[7];
    const float* mem_b2   = (const float*)d_in[8];
    const float* emb_w    = (const float*)d_in[9];
    const float* emb_b    = (const float*)d_in[10];
    const float* sa_qkv_w = (const float*)d_in[11];
    const float* sa_qkv_b = (const float*)d_in[12];
    const float* sa_o_w   = (const float*)d_in[13];
    const float* sa_o_b   = (const float*)d_in[14];
    const float* ca_qkv_w = (const float*)d_in[15];
    const float* ca_qkv_b = (const float*)d_in[16];
    const float* ca_o_w   = (const float*)d_in[17];
    const float* ca_o_b   = (const float*)d_in[18];
    const float* ln1_g    = (const float*)d_in[19];
    const float* ln1_b    = (const float*)d_in[20];
    const float* ln2_g    = (const float*)d_in[21];
    const float* ln2_b    = (const float*)d_in[22];
    const float* ln3_g    = (const float*)d_in[23];
    const float* ln3_b    = (const float*)d_in[24];
    const float* ff1_w    = (const float*)d_in[25];
    const float* ff1_b    = (const float*)d_in[26];
    const float* ff2_w    = (const float*)d_in[27];
    const float* ff2_b    = (const float*)d_in[28];
    const float* op_w1    = (const float*)d_in[29];
    const float* op_b1    = (const float*)d_in[30];
    const float* op_w2    = (const float*)d_in[31];
    const float* op_b2    = (const float*)d_in[32];
    float* out = (float*)d_out;

    // workspace carve-up (256B aligned)
    size_t off = 0;
    auto alloc = [&](size_t bytes) -> void* {
        void* p = (char*)d_ws + off;
        off += (bytes + 255) & ~(size_t)255;
        return p;
    };
    const int n_qkv = L_ * 768 * 256;
    const int n_o   = L_ * 256 * 256;
    const int n_f1  = L_ * 1024 * 256;
    const int n_f2  = L_ * 256 * 1024;
    _Float16* w_qkv16 = (_Float16*)alloc((size_t)n_qkv * 2);
    _Float16* w_o16   = (_Float16*)alloc((size_t)n_o * 2);
    _Float16* w_f1_16 = (_Float16*)alloc((size_t)n_f1 * 2);
    _Float16* w_f2_16 = (_Float16*)alloc((size_t)n_f2 * 2);
    float* ca_const = (float*)alloc((size_t)L_ * B_ * D_ * 4);
    float* kcache   = (float*)alloc((size_t)L_ * B_ * T_ * D_ * 4);
    float* vcache   = (float*)alloc((size_t)L_ * B_ * T_ * D_ * 4);

    cvt_f32_to_f16<<<(n_qkv + 255) / 256, 256, 0, stream>>>(sa_qkv_w, w_qkv16, n_qkv);
    cvt_f32_to_f16<<<(n_o   + 255) / 256, 256, 0, stream>>>(sa_o_w,   w_o16,   n_o);
    cvt_f32_to_f16<<<(n_f1  + 255) / 256, 256, 0, stream>>>(ff1_w,    w_f1_16, n_f1);
    cvt_f32_to_f16<<<(n_f2  + 255) / 256, 256, 0, stream>>>(ff2_w,    w_f2_16, n_f2);

    setup_memory_kernel<<<1, 256, 0, stream>>>(
        z_style, z_skill, mem_w1, mem_b1, mem_ln_g, mem_ln_b, mem_w2, mem_b2,
        ca_qkv_w, ca_qkv_b, ca_o_w, ca_o_b, ca_const);

    decoder_kernel<<<1, 256, 0, stream>>>(
        start_tk, emb_w, emb_b,
        w_qkv16, sa_qkv_b, w_o16, sa_o_b,
        w_f1_16, ff1_b, w_f2_16, ff2_b,
        ln1_g, ln1_b, ln2_g, ln2_b, ln3_g, ln3_b,
        ca_const, op_w1, op_b1, op_w2, op_b2,
        kcache, vcache, out);
}
